// CausalSelfAttention_37177236914665
// MI455X (gfx1250) — matrix operand order repaired and checked by reading, not compiled, not run on hardware
//
#include <hip/hip_runtime.h>

typedef __bf16 bf16_t;
typedef __attribute__((ext_vector_type(16))) __bf16 v16bf;
typedef __attribute__((ext_vector_type(8)))  float  v8f;

#define B_DIM 2
#define T_DIM 2048
#define C_DIM 1024
#define H_DIM 16
#define HD    64

// f32 -> bf16, round-to-nearest-even, branch-free (no reliance on __bf16 cvt ops)
__device__ __forceinline__ bf16_t f2bf(float f) {
  unsigned u = __builtin_bit_cast(unsigned, f);
  unsigned r = (u + 0x7FFFu + ((u >> 16) & 1u)) >> 16;
  unsigned short s = (unsigned short)r;
  return __builtin_bit_cast(bf16_t, s);
}

union FragU { v16bf v; float4 f[2]; };

// ---- CDNA5 async global->LDS copy (GV mode, 16B per lane, ASYNCcnt) ----
__device__ __forceinline__ void async_load_b128(void* lds_dst, const void* gsrc) {
  unsigned lds_off = (unsigned)(size_t)lds_dst;   // low 32 bits of LDS-aperture addr
  asm volatile("global_load_async_to_lds_b128 %0, %1, off"
               :
               : "v"(lds_off), "v"(gsrc)
               : "memory");
}
__device__ __forceinline__ void wait_async0() {
  asm volatile("s_wait_asynccnt 0" ::: "memory");
}

// A-fragment 16x32 bf16 (ISA 7.12.2): lane<16 -> M=lane, K in {0..7, 16..23};
// lane>=16 -> M=lane-16, K in {8..15, 24..31}. base = &tile[row0][0], ldm in elems.
__device__ __forceinline__ v16bf load_frag_a(const bf16_t* base, int ldm, int lane) {
  int m  = lane & 15;
  int k0 = (lane < 16) ? 0 : 8;
  const bf16_t* p = base + m * ldm + k0;
  FragU u;
  u.f[0] = *(const float4*)(p);
  u.f[1] = *(const float4*)(p + 16);
  return u.v;
}

// B-fragment 32x16 bf16 read from a TRANSPOSED tile Bt[n][k]:
// lane<16 -> N=lane, K in {0..7, 16..23}; lane>=16 -> N=lane-16, K in {8..15, 24..31}. base = &Bt[col0][0].
__device__ __forceinline__ v16bf load_frag_b(const bf16_t* base, int ldm, int lane) {
  int n  = lane & 15;
  int k0 = (lane < 16) ? 0 : 8;
  const bf16_t* p = base + n * ldm + k0;
  FragU u;
  u.f[0] = *(const float4*)(p);
  u.f[1] = *(const float4*)(p + 16);
  return u.v;
}

// ---------------------------------------------------------------------------
// GEMM: C(f32) = A * B, A is MxK (f32 or bf16), B is KxN f32 (converted to bf16).
// Block tile 128x128, 8 waves, wave tile 32x64, K-step 32.
// bf16-A path stages its tile with async global->LDS (ASYNCcnt).
// ---------------------------------------------------------------------------
template <bool ABF16>
__global__ __launch_bounds__(256) void gemm_bf16_kernel(const void* __restrict__ Ap,
                                                        const float* __restrict__ Bw,
                                                        float* __restrict__ Cout,
                                                        int M, int N, int K) {
  __shared__ __align__(16) bf16_t As[128][40];   // row-major [m][k], padded
  __shared__ __align__(16) bf16_t Bs[128][40];   // transposed  [n][k], padded

  const int tid  = threadIdx.x;
  const int lane = tid & 31;
  const int wave = tid >> 5;
  const int wr   = wave >> 1;       // 0..3 -> 32-row slice
  const int wc   = wave & 1;        // 0..1 -> 64-col slice
  const int row0 = blockIdx.y * 128;
  const int col0 = blockIdx.x * 128;

  v8f acc[2][4];
  const v8f vzero = {0.f, 0.f, 0.f, 0.f, 0.f, 0.f, 0.f, 0.f};
#pragma unroll
  for (int i = 0; i < 2; ++i)
#pragma unroll
    for (int j = 0; j < 4; ++j) acc[i][j] = vzero;

  const int arow = tid >> 1, ahalf = tid & 1;     // A loader: 2 threads / row
  const int bn = tid & 127, bkh = tid >> 7;       // B loader: 2 threads / col

  for (int k0 = 0; k0 < K; k0 += 32) {
    // ---- stage A tile (128x32) into LDS as bf16 ----
    if (ABF16) {
      const bf16_t* ga = (const bf16_t*)Ap + (size_t)(row0 + arow) * K + k0 + ahalf * 16;
      async_load_b128(&As[arow][ahalf * 16],     ga);
      async_load_b128(&As[arow][ahalf * 16 + 8], ga + 8);
    } else {
      const float* ga = (const float*)Ap + (size_t)(row0 + arow) * K + k0 + ahalf * 16;
#pragma unroll
      for (int j = 0; j < 16; j += 4) {
        float4 f = *(const float4*)(ga + j);
        As[arow][ahalf * 16 + j + 0] = f2bf(f.x);
        As[arow][ahalf * 16 + j + 1] = f2bf(f.y);
        As[arow][ahalf * 16 + j + 2] = f2bf(f.z);
        As[arow][ahalf * 16 + j + 3] = f2bf(f.w);
      }
    }
    // ---- stage B tile (32x128) transposed into LDS as bf16 ----
#pragma unroll
    for (int j = 0; j < 16; ++j) {
      float f = Bw[(size_t)(k0 + bkh * 16 + j) * N + col0 + bn];
      Bs[bn][bkh * 16 + j] = f2bf(f);
    }
    if (k0 + 32 < K) {   // prefetch next K tile -> global_prefetch_b8
      if (ABF16)
        __builtin_prefetch((const bf16_t*)Ap + (size_t)(row0 + arow) * K + k0 + 32, 0, 0);
      else
        __builtin_prefetch((const float*)Ap + (size_t)(row0 + arow) * K + k0 + 32, 0, 0);
      __builtin_prefetch(&Bw[(size_t)(k0 + 32 + bkh * 16) * N + col0 + bn], 0, 0);
    }
    if (ABF16) wait_async0();
    __syncthreads();

    v16bf af[2], bf[4];
#pragma unroll
    for (int i = 0; i < 2; ++i) af[i] = load_frag_a(&As[wr * 32 + i * 16][0], 40, lane);
#pragma unroll
    for (int j = 0; j < 4; ++j) bf[j] = load_frag_b(&Bs[wc * 64 + j * 16][0], 40, lane);
#pragma unroll
    for (int i = 0; i < 2; ++i)
#pragma unroll
      for (int j = 0; j < 4; ++j)
        acc[i][j] = __builtin_amdgcn_wmma_f32_16x16x32_bf16(
            false, af[i], false, bf[j], (short)0, acc[i][j], false, false);
    __syncthreads();
  }

  // epilogue: C layout -> VGPR r holds M = r + 8*(lane>=16), N = lane&15
  const int mhi = (lane >> 4) << 3;
  const int nl  = lane & 15;
#pragma unroll
  for (int i = 0; i < 2; ++i)
#pragma unroll
    for (int j = 0; j < 4; ++j)
#pragma unroll
      for (int r = 0; r < 8; ++r) {
        int gm = row0 + wr * 32 + i * 16 + r + mhi;
        int gn = col0 + wc * 64 + j * 16 + nl;
        Cout[(size_t)gm * N + gn] = acc[i][j][r];
      }
}

// ---------------------------------------------------------------------------
// RoPE + reshape. qkv f32 (B,T,3C) -> k,v f32 (B,H,T,hd) to d_out;
// bf16 q (scaled 1/sqrt(hd)), k, v to workspace in (B*H, T, hd).
// ---------------------------------------------------------------------------
__global__ __launch_bounds__(256) void rope_qkv_kernel(const float* __restrict__ qkv,
                                                       bf16_t* __restrict__ qb,
                                                       bf16_t* __restrict__ kb,
                                                       bf16_t* __restrict__ vb,
                                                       float* __restrict__ kout,
                                                       float* __restrict__ vout) {
  int tid = blockIdx.x * 256 + threadIdx.x;       // B*H*T*32 threads
  int i = tid & 31;
  int t = (tid >> 5) & (T_DIM - 1);
  int h = (tid >> 16) & (H_DIM - 1);
  int b = tid >> 20;

  size_t row = (size_t)(b * T_DIM + t) * (3 * C_DIM);
  int cq = h * HD + i;
  float q1 = qkv[row + cq],               q2 = qkv[row + cq + 32];
  float k1 = qkv[row + C_DIM + cq],       k2 = qkv[row + C_DIM + cq + 32];
  float v1 = qkv[row + 2 * C_DIM + cq],   v2 = qkv[row + 2 * C_DIM + cq + 32];

  // inv_freq = 10000^(-i/32) = exp(-i * ln(10000)/32)
  float ang = (float)t * __expf(-(float)i * (9.210340371976184f / 32.0f));
  float c = __cosf(ang), s = __sinf(ang);
  float qr1 = q1 * c - q2 * s, qr2 = q2 * c + q1 * s;
  float kr1 = k1 * c - k2 * s, kr2 = k2 * c + k1 * s;

  size_t o = (size_t)((b * H_DIM + h) * T_DIM + t) * HD + i;
  kout[o] = kr1;  kout[o + 32] = kr2;
  vout[o] = v1;   vout[o + 32] = v2;
  qb[o] = f2bf(qr1 * 0.125f);  qb[o + 32] = f2bf(qr2 * 0.125f);  // fold 1/sqrt(64)
  kb[o] = f2bf(kr1);           kb[o + 32] = f2bf(kr2);
  vb[o] = f2bf(v1);            vb[o + 32] = f2bf(v2);
}

// ---------------------------------------------------------------------------
// Flash attention: grid (T/64, B*H), 128 threads = 4 waves, wave = 16 q-rows.
// K tile staged with async global->LDS; V tile transposed through registers.
// ---------------------------------------------------------------------------
__global__ __launch_bounds__(128) void flash_attn_kernel(const bf16_t* __restrict__ Q,
                                                         const bf16_t* __restrict__ Kc,
                                                         const bf16_t* __restrict__ Vc,
                                                         bf16_t* __restrict__ O) {
  __shared__ __align__(16) bf16_t Ks[64][72];      // K tile, row-major [k][d]
  __shared__ __align__(16) bf16_t Vt[64][72];      // V tile, transposed [d][k]
  __shared__ __align__(16) bf16_t Ps[4][16][72];   // P staging per wave

  const int tid  = threadIdx.x;
  const int lane = tid & 31;
  const int w    = tid >> 5;
  const int bh   = blockIdx.y;
  const int qt   = blockIdx.x;
  const int q0w  = qt * 64 + w * 16;

  // resident Q A-fragments (16 rows x 64 d = 2 chunks of K=32)
  const bf16_t* qp = Q + ((size_t)bh * T_DIM + q0w) * HD;
  v16bf qf[2];
  {
    int m = lane & 15, k0 = (lane < 16) ? 0 : 8;
#pragma unroll
    for (int c = 0; c < 2; ++c) {
      FragU u;
      const bf16_t* p = qp + m * HD + 32 * c + k0;
      u.f[0] = *(const float4*)(p);
      u.f[1] = *(const float4*)(p + 16);
      qf[c] = u.v;
    }
  }

  const v8f vzero = {0.f, 0.f, 0.f, 0.f, 0.f, 0.f, 0.f, 0.f};
  float mrun[8], lsum[8];
  v8f o[4];
#pragma unroll
  for (int r = 0; r < 8; ++r) { mrun[r] = -3.0e38f; lsum[r] = 0.f; }
#pragma unroll
  for (int n = 0; n < 4; ++n) o[n] = vzero;

  const int lrow = tid >> 1, lhalf = tid & 1;      // K/V loaders: 2 threads / row

  for (int jt = 0; jt <= qt; ++jt) {
    int j0 = jt * 64;
    const bf16_t* kp = Kc + ((size_t)bh * T_DIM + j0 + lrow) * HD + lhalf * 32;
    const bf16_t* vp = Vc + ((size_t)bh * T_DIM + j0 + lrow) * HD + lhalf * 32;
    // K tile: async DMA into LDS (no conversion needed), ASYNCcnt tracked
#pragma unroll
    for (int c2 = 0; c2 < 4; ++c2)
      async_load_b128(&Ks[lrow][lhalf * 32 + c2 * 8], kp + c2 * 8);
    // V tile: load + transpose-scatter into Vt
#pragma unroll
    for (int c2 = 0; c2 < 4; ++c2) {
      union { float4 f; bf16_t h[8]; } uv;
      uv.f = *(const float4*)(vp + c2 * 8);
#pragma unroll
      for (int e = 0; e < 8; ++e) Vt[lhalf * 32 + c2 * 8 + e][lrow] = uv.h[e];
    }
    wait_async0();
    __syncthreads();

    // S = Q K^T  (B-frag from row-major K: lane = key column, contiguous d)
    v8f s[4];
#pragma unroll
    for (int n = 0; n < 4; ++n) s[n] = vzero;
    {
      int nl = lane & 15, kb0 = (lane < 16) ? 0 : 8;
#pragma unroll
      for (int c = 0; c < 2; ++c)
#pragma unroll
        for (int n = 0; n < 4; ++n) {
          FragU u;
          const bf16_t* p = &Ks[n * 16 + nl][32 * c + kb0];
          u.f[0] = *(const float4*)(p);
          u.f[1] = *(const float4*)(p + 16);
          s[n] = __builtin_amdgcn_wmma_f32_16x16x32_bf16(
              false, qf[c], false, u.v, (short)0, s[n], false, false);
        }
    }

    if (jt == qt) {  // causal mask only needed on the diagonal tile
      int nl = lane & 15, mhi = (lane >> 4) << 3;
#pragma unroll
      for (int n = 0; n < 4; ++n)
#pragma unroll
        for (int r = 0; r < 8; ++r)
          if (j0 + n * 16 + nl > q0w + r + mhi) s[n][r] = -3.0e38f;
    }

    // online softmax (row stats replicated across 16-lane halves)
    float tmax[8];
#pragma unroll
    for (int r = 0; r < 8; ++r) {
      float v = fmaxf(fmaxf(s[0][r], s[1][r]), fmaxf(s[2][r], s[3][r]));
      v = fmaxf(v, __shfl_xor(v, 1));
      v = fmaxf(v, __shfl_xor(v, 2));
      v = fmaxf(v, __shfl_xor(v, 4));
      v = fmaxf(v, __shfl_xor(v, 8));
      tmax[r] = v;
    }
#pragma unroll
    for (int r = 0; r < 8; ++r) {
      float mnew  = fmaxf(mrun[r], tmax[r]);
      float alpha = __expf(mrun[r] - mnew);
      mrun[r] = mnew;
      lsum[r] *= alpha;
#pragma unroll
      for (int n = 0; n < 4; ++n) o[n][r] *= alpha;
      float rs = 0.f;
#pragma unroll
      for (int n = 0; n < 4; ++n) {
        float p = __expf(s[n][r] - mnew);
        s[n][r] = p;
        rs += p;
      }
      rs += __shfl_xor(rs, 1); rs += __shfl_xor(rs, 2);
      rs += __shfl_xor(rs, 4); rs += __shfl_xor(rs, 8);
      lsum[r] += rs;
    }

    // stage P (C-layout) -> LDS, re-read as A-fragments
    {
      int nl = lane & 15, mhi = (lane >> 4) << 3;
#pragma unroll
      for (int n = 0; n < 4; ++n)
#pragma unroll
        for (int r = 0; r < 8; ++r)
          Ps[w][r + mhi][n * 16 + nl] = f2bf(s[n][r]);
    }

    // O += P V  (B-frag from transposed Vt: lane = d column, contiguous k)
    {
      int m = lane & 15, ka0 = (lane < 16) ? 0 : 8;
      int kb0 = (lane < 16) ? 0 : 8;
#pragma unroll
      for (int c = 0; c < 2; ++c) {
        FragU up;
        const bf16_t* pp = &Ps[w][m][32 * c + ka0];
        up.f[0] = *(const float4*)(pp);
        up.f[1] = *(const float4*)(pp + 16);
#pragma unroll
        for (int n = 0; n < 4; ++n) {
          FragU uv;
          const bf16_t* vbp = &Vt[n * 16 + m][32 * c + kb0];
          uv.f[0] = *(const float4*)(vbp);
          uv.f[1] = *(const float4*)(vbp + 16);
          o[n] = __builtin_amdgcn_wmma_f32_16x16x32_bf16(
              false, up.v, false, uv.v, (short)0, o[n], false, false);
        }
      }
    }
    __syncthreads();
  }

  // epilogue: write bf16 attention output in (B,T,C) layout for the out-proj GEMM
  int b = bh >> 4, h = bh & 15;
  int nl = lane & 15, mhi = (lane >> 4) << 3;
#pragma unroll
  for (int r = 0; r < 8; ++r) {
    float inv = 1.0f / lsum[r];
    int q = q0w + r + mhi;
    bf16_t* op = O + (size_t)(b * T_DIM + q) * C_DIM + h * HD;
#pragma unroll
    for (int n = 0; n < 4; ++n) op[n * 16 + nl] = f2bf(o[n][r] * inv);
  }
}

// ---------------------------------------------------------------------------
extern "C" void kernel_launch(void* const* d_in, const int* in_sizes, int n_in,
                              void* d_out, int out_size, void* d_ws, size_t ws_size,
                              hipStream_t stream) {
  const float* x     = (const float*)d_in[0];   // (B,T,C)
  const float* w_qkv = (const float*)d_in[1];   // (C,3C)
  const float* w_out = (const float*)d_in[2];   // (C,C)

  float* out   = (float*)d_out;                 // (B,T,C)
  float* k_out = out + (size_t)B_DIM * T_DIM * C_DIM;
  float* v_out = k_out + (size_t)B_DIM * T_DIM * C_DIM;

  char* ws = (char*)d_ws;
  float*  qkv  = (float*)ws;                                 // 4096*3072 f32 = 48 MB
  bf16_t* qb   = (bf16_t*)(ws + (size_t)4096 * 3072 * 4);
  bf16_t* kb   = qb + (size_t)B_DIM * H_DIM * T_DIM * HD;
  bf16_t* vb   = kb + (size_t)B_DIM * H_DIM * T_DIM * HD;
  bf16_t* attn = vb + (size_t)B_DIM * H_DIM * T_DIM * HD;

  const int M = B_DIM * T_DIM;   // 4096

  // 1) qkv = x @ w_qkv
  gemm_bf16_kernel<false><<<dim3((3 * C_DIM) / 128, M / 128), 256, 0, stream>>>(
      (const void*)x, w_qkv, qkv, M, 3 * C_DIM, C_DIM);

  // 2) RoPE + reshape, emit f32 k/v outputs and bf16 q/k/v
  rope_qkv_kernel<<<(B_DIM * H_DIM * T_DIM * 32) / 256, 256, 0, stream>>>(
      qkv, qb, kb, vb, k_out, v_out);

  // 3) flash attention -> bf16 (B,T,C)
  flash_attn_kernel<<<dim3(T_DIM / 64, B_DIM * H_DIM), 128, 0, stream>>>(qb, kb, vb, attn);

  // 4) out = attn @ w_out
  gemm_bf16_kernel<true><<<dim3(C_DIM / 128, M / 128), 256, 0, stream>>>(
      (const void*)attn, w_out, out, M, C_DIM, C_DIM);
}
